// AGNNFeatureExtractor_11811160064580
// MI455X (gfx1250) — matrix operand order repaired and checked by reading, not compiled, not run on hardware
//
#include <hip/hip_runtime.h>
#include <hip/hip_bf16.h>
#include <stdint.h>

// ---------------------------------------------------------------------------
// AGNN feature extractor for gfx1250 (MI455X, wave32, WMMA + TDM/async-LDS).
// ---------------------------------------------------------------------------

typedef __attribute__((ext_vector_type(16))) _Float16 v16h;
typedef __attribute__((ext_vector_type(8)))  float    v8f;
typedef __attribute__((ext_vector_type(4)))  unsigned int u32x4;
typedef __attribute__((ext_vector_type(4)))  int      i32x4;
typedef __attribute__((ext_vector_type(8)))  int      i32x8;
typedef unsigned int u32;

#define BB 256
#define NNODE 400
#define E1 1520
#define ETOT 1920
#define ATTN_SCALE 0.17677669529663687f  // 1/sqrt(32)

union Frag { v16h h; u32 u[8]; _Float16 e[16]; };

__device__ inline v8f zero8() { v8f z; for (int i = 0; i < 8; ++i) z[i] = 0.f; return z; }

// 16-bit A-matrix 16x32 fragment K index for vgpr-pair i (ISA 7.12.2)
__device__ inline int a_k(int i, int lane) {
  int koff = (lane & 16) ? 8 : 0;
  return (i < 4) ? (koff + 2 * i) : (16 + koff + 2 * (i - 4));
}

// ---------------------------------------------------------------------------
// Tensor Data Mover: 2-D tile load (f16 elements) into LDS (ISA ch.8 D#).
// group0: count=1 | lds_addr | global_addr | type=2("image")
// group1: data_size=1(2B), tensor_dim0/1, tile_dim0/1, tensor_dim0_stride
// ---------------------------------------------------------------------------
__device__ inline void tdm_load_2d_f16(unsigned lds_addr, const void* gaddr,
                                       unsigned tile_d0, unsigned tile_d1,
                                       unsigned long long tensor_d0,
                                       unsigned long long tensor_d1,
                                       unsigned long long stride0) {
  unsigned long long ga = (unsigned long long)(uintptr_t)gaddr;
  u32x4 g0;
  g0[0] = 1u;                                            // count=1, user D#
  g0[1] = lds_addr;                                      // lds_addr[31:0]
  g0[2] = (unsigned)(ga & 0xffffffffu);                  // global_addr low
  g0[3] = (unsigned)((ga >> 32) & 0x01ffffffu) | (2u << 30);  // addr hi | type=2
  i32x8 g1;
  g1[0] = (int)(1u << 16);                               // data_size=1 -> 2 bytes
  g1[1] = (int)((unsigned)(tensor_d0 & 0xffffu) << 16);  // tensor_dim0[15:0]
  g1[2] = (int)((unsigned)((tensor_d0 >> 16) & 0xffffu) |
                ((unsigned)(tensor_d1 & 0xffffu) << 16));
  g1[3] = (int)((unsigned)((tensor_d1 >> 16) & 0xffffu) | (tile_d0 << 16));
  g1[4] = (int)(tile_d1 & 0xffffu);                      // tile_dim1 (tile_dim2=0)
  g1[5] = (int)(unsigned)(stride0 & 0xffffffffu);        // dim0 stride low
  g1[6] = (int)(unsigned)((stride0 >> 32) & 0xffffu);    // stride hi | dim1stride=0
  g1[7] = 0;
  i32x4 z4 = {0, 0, 0, 0};
#if defined(__clang_major__) && (__clang_major__ >= 23)
  i32x8 z8 = {0, 0, 0, 0, 0, 0, 0, 0};
  __builtin_amdgcn_tensor_load_to_lds(g0, g1, z4, z4, z8, 0);
#else
  __builtin_amdgcn_tensor_load_to_lds(g0, g1, z4, z4, 0);
#endif
}

// ---------------------------------------------------------------------------
// Generic WMMA GEMM (manual staging): C = act(A@W + bias), K%32==0, N%64==0.
// Block 256 thr = 8 waves, block tile 128x64, wave tile 32x32 (2x2 wmma).
// ---------------------------------------------------------------------------
template<int RELU>
__global__ void __launch_bounds__(256)
gemm_kernel(const _Float16* __restrict__ A, const _Float16* __restrict__ W,
            const float* __restrict__ bias, float* __restrict__ C32,
            _Float16* __restrict__ C16, int M, int N, int K)
{
  __shared__ __align__(16) _Float16 As[128 * 32];   // row-major [m][k]
  __shared__ __align__(16) _Float16 BsT[64 * 32];   // transposed [n][k]
  const int tid  = threadIdx.x;
  const int lane = tid & 31;
  const int wave = tid >> 5;
  const int wm = wave >> 1, wn = wave & 1;
  const int m0 = blockIdx.x * 128;
  const int n0 = blockIdx.y * 64;

  v8f acc[2][2];
  for (int i = 0; i < 2; ++i) for (int j = 0; j < 2; ++j) acc[i][j] = zero8();

  for (int kt = 0; kt < K; kt += 32) {
    for (int t = 0; t < 16; ++t) {          // 128x32 f16 A tile
      int e = tid + 256 * t;
      int m = e >> 5, k = e & 31;
      int gr = m0 + m;
      As[e] = (gr < M) ? A[(size_t)gr * K + kt + k] : (_Float16)0.f;
    }
    for (int t = 0; t < 8; ++t) {           // 32x64 W tile, stored transposed
      int e = tid + 256 * t;
      int k = e >> 6, n = e & 63;
      BsT[n * 32 + k] = W[(size_t)(kt + k) * N + n0 + n];
    }
    if (kt + 32 < K)
      __builtin_prefetch(&W[(size_t)(kt + 32) * N + n0 + (tid & 63)], 0, 1);
    __syncthreads();

    Frag af[2], bf[2];
    const int koff = (lane & 16) ? 8 : 0;
    for (int ti = 0; ti < 2; ++ti) {
      const _Float16* ap = &As[(wm * 32 + ti * 16 + (lane & 15)) * 32];
      af[ti].u[0] = *(const u32*)&ap[koff + 0];
      af[ti].u[1] = *(const u32*)&ap[koff + 2];
      af[ti].u[2] = *(const u32*)&ap[koff + 4];
      af[ti].u[3] = *(const u32*)&ap[koff + 6];
      af[ti].u[4] = *(const u32*)&ap[16 + koff + 0];
      af[ti].u[5] = *(const u32*)&ap[16 + koff + 2];
      af[ti].u[6] = *(const u32*)&ap[16 + koff + 4];
      af[ti].u[7] = *(const u32*)&ap[16 + koff + 6];
    }
    const int kb = (lane & 16) ? 16 : 0;
    for (int tj = 0; tj < 2; ++tj) {
      const _Float16* bp = &BsT[(wn * 32 + tj * 16 + (lane & 15)) * 32];
      for (int i = 0; i < 8; ++i) bf[tj].u[i] = *(const u32*)&bp[a_k(i, lane)];
    }
    for (int ti = 0; ti < 2; ++ti)
      for (int tj = 0; tj < 2; ++tj)
        acc[ti][tj] = __builtin_amdgcn_wmma_f32_16x16x32_f16(
            false, af[ti].h, false, bf[tj].h, (short)0, acc[ti][tj], false, false);
    __syncthreads();
  }

  for (int ti = 0; ti < 2; ++ti) {
    for (int tj = 0; tj < 2; ++tj) {
      int col = n0 + wn * 32 + tj * 16 + (lane & 15);
      float bv = bias ? bias[col] : 0.f;
      for (int i = 0; i < 8; ++i) {
        int row = m0 + wm * 32 + ti * 16 + i + ((lane & 16) ? 8 : 0);
        if (row < M) {
          float v = acc[ti][tj][i] + bv;
          if (RELU) v = v > 0.f ? v : 0.f;
          size_t o = (size_t)row * N + col;
          if (C16) C16[o] = (_Float16)v;
          if (C32) C32[o] = v;
        }
      }
    }
  }
}

// ---------------------------------------------------------------------------
// WMMA GEMM with DMA staging: A tile via Tensor Data Mover (TENSORcnt),
// W tile via per-lane global_load_async_to_lds_b128 (ASYNCcnt).
// ---------------------------------------------------------------------------
template<int RELU>
__global__ void __launch_bounds__(256)
gemm_async_kernel(const _Float16* __restrict__ A, const _Float16* __restrict__ W,
                  const float* __restrict__ bias, float* __restrict__ C32,
                  _Float16* __restrict__ C16, int M, int N, int K)
{
  __shared__ __align__(16) _Float16 As[128 * 32];   // row-major [m][k]  (TDM dest)
  __shared__ __align__(16) _Float16 Bs[32 * 64];    // row-major [k][n]  (async dest)
  const int tid  = threadIdx.x;
  const int lane = tid & 31;
  const int wave = tid >> 5;
  const int wm = wave >> 1, wn = wave & 1;
  const int m0 = blockIdx.x * 128;
  const int n0 = blockIdx.y * 64;

  v8f acc[2][2];
  for (int i = 0; i < 2; ++i) for (int j = 0; j < 2; ++j) acc[i][j] = zero8();

  // per-thread async W-tile slice: 32x64 f16 = 4KB = 256 x 16B
  const int bk = tid >> 3;            // 0..31
  const int bn = (tid & 7) * 8;       // 0..56
  const unsigned bs_lds = (unsigned)(uintptr_t)&Bs[bk * 64 + bn];
  const unsigned as_lds = (unsigned)(uintptr_t)&As[0];

  for (int kt = 0; kt < K; kt += 32) {
    if (wave == 0) {
      // TDM: 128 lines x 32 f16, line stride K elements; rows >= M-m0 read 0
      tdm_load_2d_f16(as_lds, A + (size_t)m0 * K + kt,
                      /*tile_d0=*/32u, /*tile_d1=*/128u,
                      /*tensor_d0=*/(unsigned long long)K,
                      /*tensor_d1=*/(unsigned long long)(M - m0),
                      /*stride0=*/(unsigned long long)K);
    }
    {
      unsigned long long ga =
          (unsigned long long)(uintptr_t)(W + (size_t)(kt + bk) * N + n0 + bn);
      asm volatile("global_load_async_to_lds_b128 %0, %1, off"
                   :: "v"(bs_lds), "v"(ga) : "memory");
    }
    asm volatile("s_wait_asynccnt 0x0" ::: "memory");
    if (wave == 0) __builtin_amdgcn_s_wait_tensorcnt((short)0);
    __syncthreads();

    Frag af[2], bf[2];
    const int koff = (lane & 16) ? 8 : 0;
    for (int ti = 0; ti < 2; ++ti) {
      const _Float16* ap = &As[(wm * 32 + ti * 16 + (lane & 15)) * 32];
      af[ti].u[0] = *(const u32*)&ap[koff + 0];
      af[ti].u[1] = *(const u32*)&ap[koff + 2];
      af[ti].u[2] = *(const u32*)&ap[koff + 4];
      af[ti].u[3] = *(const u32*)&ap[koff + 6];
      af[ti].u[4] = *(const u32*)&ap[16 + koff + 0];
      af[ti].u[5] = *(const u32*)&ap[16 + koff + 2];
      af[ti].u[6] = *(const u32*)&ap[16 + koff + 4];
      af[ti].u[7] = *(const u32*)&ap[16 + koff + 6];
    }
    const int kb = (lane & 16) ? 16 : 0;
    for (int tj = 0; tj < 2; ++tj) {
      int n = wn * 32 + tj * 16 + (lane & 15);
      for (int i = 0; i < 8; ++i) {
        bf[tj].e[2 * i]     = Bs[(a_k(i, lane))     * 64 + n];
        bf[tj].e[2 * i + 1] = Bs[(a_k(i, lane) + 1) * 64 + n];
      }
    }
    for (int ti = 0; ti < 2; ++ti)
      for (int tj = 0; tj < 2; ++tj)
        acc[ti][tj] = __builtin_amdgcn_wmma_f32_16x16x32_f16(
            false, af[ti].h, false, bf[tj].h, (short)0, acc[ti][tj], false, false);
    __syncthreads();
  }

  for (int ti = 0; ti < 2; ++ti) {
    for (int tj = 0; tj < 2; ++tj) {
      int col = n0 + wn * 32 + tj * 16 + (lane & 15);
      float bv = bias ? bias[col] : 0.f;
      for (int i = 0; i < 8; ++i) {
        int row = m0 + wm * 32 + ti * 16 + i + ((lane & 16) ? 8 : 0);
        if (row < M) {
          float v = acc[ti][tj][i] + bv;
          if (RELU) v = v > 0.f ? v : 0.f;
          size_t o = (size_t)row * N + col;
          if (C16) C16[o] = (_Float16)v;
          if (C32) C32[o] = v;
        }
      }
    }
  }
}

// ---------------------------------------------------------------------------
// Fused multi-head cross-attention: one block = (batch b, 16-query tile);
// 4 waves = 4 heads. K/V (65 keys x 32 dh per head) + scores live in LDS.
// S = Q@K^T via WMMA, softmax in LDS, O = P@V via WMMA.
// Object row layout: [0,256)=main(b), [256,13056)=agv(b*50+j), then stat.
// ---------------------------------------------------------------------------
__global__ void __launch_bounds__(128)
attn_kernel(const _Float16* __restrict__ Q, const _Float16* __restrict__ Kg,
            const _Float16* __restrict__ Vg, const float* __restrict__ agvs,
            const float* __restrict__ stat, const float* __restrict__ nodes,
            _Float16* __restrict__ O)
{
  __shared__ __align__(16) _Float16 Ks[4 * 65 * 32];
  __shared__ __align__(16) _Float16 Vs[4 * 65 * 32];
  __shared__ __align__(16) float    Sld[4 * 16 * 96];

  const int tid  = threadIdx.x;
  const int lane = tid & 31;
  const int h    = tid >> 5;
  const int qt   = blockIdx.x;
  const int b    = blockIdx.y;

  for (int i = tid; i < 65 * 64; i += 128) {   // 65 rows x 64 uint-pairs of K and V
    int key = i >> 6;
    int cp  = i & 63;
    int row;
    if (key == 0)       row = b;
    else if (key <= 50) row = 256 + b * 50 + (key - 1);
    else                row = 256 + 12800 + b * 14 + (key - 51);
    u32 kv = *(const u32*)(Kg + (size_t)row * 128 + cp * 2);
    u32 vv = *(const u32*)(Vg + (size_t)row * 128 + cp * 2);
    int hh = cp >> 4, dhp = (cp & 15) * 2;
    *(u32*)&Ks[(hh * 65 + key) * 32 + dhp] = kv;
    *(u32*)&Vs[(hh * 65 + key) * 32 + dhp] = vv;
  }
  __syncthreads();

  Frag qf;                                     // Q fragment 16x32
  {
    const _Float16* qp = Q + (size_t)(b * 400 + qt * 16 + (lane & 15)) * 128 + h * 32;
    for (int i = 0; i < 8; ++i) qf.u[i] = *(const u32*)(qp + a_k(i, lane));
  }

  for (int kt = 0; kt < 5; ++kt) {             // 5 key tiles of 16 (65 keys, pad 80)
    Frag bfr;
    {
      int key = kt * 16 + (lane & 15);
      int kb = (lane & 16) ? 16 : 0;
      for (int i = 0; i < 8; ++i)
        bfr.u[i] = (key < 65) ? *(const u32*)&Ks[(h * 65 + key) * 32 + a_k(i, lane)] : 0u;
    }
    v8f s = __builtin_amdgcn_wmma_f32_16x16x32_f16(false, qf.h, false, bfr.h,
                                                   (short)0, zero8(), false, false);
    int mk = kt * 16 + (lane & 15);
    int rbase = (lane & 16) ? 8 : 0;
    for (int i = 0; i < 8; ++i) {
      int r  = i + rbase;
      int nq = qt * 16 + r;
      float sc = s[i] * ATTN_SCALE;
      bool masked;
      if (mk >= 65)       masked = true;
      else if (mk == 0)   masked = false;
      else {
        float ox, oy;
        if (mk < 51) { ox = agvs[((size_t)b * 51 + mk) * 40 + 4];
                       oy = agvs[((size_t)b * 51 + mk) * 40 + 5]; }
        else         { ox = stat[((size_t)b * 14 + (mk - 51)) * 40 + 0];
                       oy = stat[((size_t)b * 14 + (mk - 51)) * 40 + 1]; }
        float nx = nodes[2 * nq], ny = nodes[2 * nq + 1];
        bool cx = fabsf(ox - nx) <= 1e-8f + 0.05f * fabsf(nx);
        bool cy = fabsf(oy - ny) <= 1e-8f + 0.05f * fabsf(ny);
        masked = !(cx && cy);
      }
      Sld[(h * 16 + r) * 96 + mk] = masked ? -1.0e9f : sc;
    }
  }
  __syncthreads();

  if (lane < 16) {                             // softmax over 65 keys, one row/lane
    float* sr = &Sld[(h * 16 + lane) * 96];
    float mx = -3.0e38f;
    for (int m = 0; m < 65; ++m) mx = fmaxf(mx, sr[m]);
    float sum = 0.f;
    for (int m = 0; m < 65; ++m) { float e2 = expf(sr[m] - mx); sr[m] = e2; sum += e2; }
    float inv = 1.f / sum;
    for (int m = 0; m < 65; ++m) sr[m] *= inv;
    for (int m = 65; m < 96; ++m) sr[m] = 0.f;
  }
  __syncthreads();

  v8f oacc[2] = { zero8(), zero8() };          // O = P(16x96) @ V(96x32)
  for (int c = 0; c < 3; ++c) {
    Frag afr;
    {
      const float* sr = &Sld[(h * 16 + (lane & 15)) * 96 + c * 32];
      for (int i = 0; i < 8; ++i) {
        int k = a_k(i, lane);
        afr.e[2 * i]     = (_Float16)sr[k];
        afr.e[2 * i + 1] = (_Float16)sr[k + 1];
      }
    }
    int kb = (lane & 16) ? 16 : 0;
    for (int nj = 0; nj < 2; ++nj) {
      Frag bfr;
      int n = nj * 16 + (lane & 15);
      for (int i = 0; i < 8; ++i) {
        int k0 = c * 32 + a_k(i, lane);
        bfr.e[2 * i]     = (k0     < 65) ? Vs[(h * 65 + k0    ) * 32 + n] : (_Float16)0.f;
        bfr.e[2 * i + 1] = (k0 + 1 < 65) ? Vs[(h * 65 + k0 + 1) * 32 + n] : (_Float16)0.f;
      }
      oacc[nj] = __builtin_amdgcn_wmma_f32_16x16x32_f16(false, afr.h, false, bfr.h,
                                                        (short)0, oacc[nj], false, false);
    }
  }

  int rbase = (lane & 16) ? 8 : 0;
  for (int nj = 0; nj < 2; ++nj) {
    int col = h * 32 + nj * 16 + (lane & 15);
    for (int i = 0; i < 8; ++i) {
      int qrow = qt * 16 + i + rbase;
      O[(size_t)(b * 400 + qrow) * 128 + col] = (_Float16)oacc[nj][i];
    }
  }
}

// ---------------------------------------------------------------------------
// Elementwise / reduction / scatter kernels
// ---------------------------------------------------------------------------
__global__ void pack_w_kernel(const float* __restrict__ src, _Float16* __restrict__ dst,
                              int K, int N, int Kp) {
  int i = blockIdx.x * 256 + threadIdx.x;
  if (i >= Kp * N) return;
  int k = i / N, n = i % N;
  dst[i] = (k < K) ? (_Float16)src[(size_t)k * N + n] : (_Float16)0.f;
}

// rows: [0,256)=main(b), [256,13056)=agv, [13056,16640)=stat; 64 padded cols
__global__ void encode_obj_kernel(const float* __restrict__ agvs, const float* __restrict__ stat,
                                  const float* __restrict__ posw, const float* __restrict__ posb,
                                  _Float16* __restrict__ enc) {
  int idx = blockIdx.x * 256 + threadIdx.x;
  if (idx >= 16640 * 64) return;
  int r = idx >> 6, c = idx & 63;
  const float* obs; int type;
  if (r < 256)        { obs = agvs + (size_t)r * 51 * 40; type = 0; }
  else if (r < 13056) { int rr = r - 256; int b = rr / 50, j = rr % 50;
                        obs = agvs + ((size_t)b * 51 + 1 + j) * 40; type = 0; }
  else                { int rr = r - 13056; int b = rr / 14, s2 = rr % 14;
                        obs = stat + ((size_t)b * 14 + s2) * 40; type = 1; }
  float v = 0.f;
  if (c < 40) v = obs[c];
  else {
    int npos = (type == 0) ? 7 : 1;
    int p = (c - 40) >> 1, comp = (c - 40) & 1;
    if (p < npos) {
      int col = (type == 0) ? (2 + 2 * p) : 0;
      v = obs[col] * posw[comp] + obs[col + 1] * posw[2 + comp] + posb[comp];
    }
  }
  enc[idx] = (_Float16)v;
}

__global__ void encode_node_kernel(const float* __restrict__ nodes, const float* __restrict__ posw,
                                   const float* __restrict__ posb, _Float16* __restrict__ enc) {
  int i = blockIdx.x * 256 + threadIdx.x;
  if (i >= 416 * 32) return;
  int r = i >> 5, c = i & 31;
  float v = 0.f;
  if (r < 400) {
    if (c < 2) v = nodes[2 * r + c];
    else if (c < 4) {
      int comp = c - 2;
      v = nodes[2 * r] * posw[comp] + nodes[2 * r + 1] * posw[2 + comp] + posb[comp];
    }
  }
  enc[i] = (_Float16)v;
}

__global__ void broadcast_kernel(const _Float16* __restrict__ ne, float* __restrict__ T32,
                                 _Float16* __restrict__ T16) {
  size_t i = (size_t)blockIdx.x * 256 + threadIdx.x;
  if (i >= (size_t)BB * 400 * 128) return;
  int rd = (int)(i % (400 * 128));
  _Float16 hv = ne[rd];
  T32[i] = (float)hv;
  T16[i] = hv;
}

__global__ void __launch_bounds__(256)
add_ln_kernel(const float* __restrict__ X, const float* __restrict__ Y,
              const float* __restrict__ g, const float* __restrict__ bta,
              float* __restrict__ O32, _Float16* __restrict__ O16, int rows) {
  int lane = threadIdx.x & 31;
  int row = blockIdx.x * 8 + (threadIdx.x >> 5);
  if (row >= rows) return;
  size_t base = (size_t)row * 128 + lane * 4;
  float4 xv = *(const float4*)(X + base);
  float4 yv = *(const float4*)(Y + base);
  float a0 = xv.x + yv.x, a1 = xv.y + yv.y, a2 = xv.z + yv.z, a3 = xv.w + yv.w;
  float s = a0 + a1 + a2 + a3;
  for (int m = 1; m < 32; m <<= 1) s += __shfl_xor(s, m, 32);
  float mean = s * (1.f / 128.f);
  float c0 = a0 - mean, c1 = a1 - mean, c2 = a2 - mean, c3 = a3 - mean;
  float q = c0 * c0 + c1 * c1 + c2 * c2 + c3 * c3;
  for (int m = 1; m < 32; m <<= 1) q += __shfl_xor(q, m, 32);
  float rs = rsqrtf(q * (1.f / 128.f) + 1e-5f);
  float4 gv = *(const float4*)(g + lane * 4);
  float4 bv = *(const float4*)(bta + lane * 4);
  float o0 = gv.x * c0 * rs + bv.x, o1 = gv.y * c1 * rs + bv.y;
  float o2 = gv.z * c2 * rs + bv.z, o3 = gv.w * c3 * rs + bv.w;
  float4 ov; ov.x = o0; ov.y = o1; ov.z = o2; ov.w = o3;
  *(float4*)(O32 + base) = ov;
  O16[base + 0] = (_Float16)o0; O16[base + 1] = (_Float16)o1;
  O16[base + 2] = (_Float16)o2; O16[base + 3] = (_Float16)o3;
}

__global__ void __launch_bounds__(128)
graph_norm_kernel(float* __restrict__ X, _Float16* __restrict__ X16,
                  const float* __restrict__ gatbias, const float* __restrict__ w,
                  const float* __restrict__ gb, const float* __restrict__ ms) {
  int b = blockIdx.x, d = threadIdx.x;
  float bias = gatbias[d];
  float sum = 0.f, sq = 0.f;
  for (int n = 0; n < 400; ++n) {
    float v = X[((size_t)b * 400 + n) * 128 + d] + bias;
    sum += v; sq += v * v;
  }
  float mean = sum * (1.f / 400.f);
  float c = ms[d] * mean;
  float var = sq * (1.f / 400.f) - 2.f * c * mean + c * c;
  float rs = rsqrtf(var + 1e-5f);
  float wd = w[d], gbd = gb[d];
  for (int n = 0; n < 400; ++n) {
    size_t o = ((size_t)b * 400 + n) * 128 + d;
    float v = X[o] + bias - c;
    float y = wd * v * rs + gbd;
    X[o] = y; X16[o] = (_Float16)y;
  }
}

__device__ inline void edge_sd(const int* ei, int e, int& s, int& d) {
  if (e < E1) { s = ei[e]; d = ei[E1 + e]; }
  else        { s = e - E1; d = e - E1; }
}

__global__ void __launch_bounds__(256)
gat_edge1_kernel(const float* __restrict__ XL, const float* __restrict__ XR,
                 const float* __restrict__ att, const int* __restrict__ ei,
                 float* __restrict__ EVAL, float* __restrict__ EMAX) {
  int lane = threadIdx.x & 31;
  int id = blockIdx.x * 8 + (threadIdx.x >> 5);
  if (id >= BB * ETOT) return;
  int b = id / ETOT, e = id % ETOT;
  int s, d; edge_sd(ei, e, s, d);
  const float* xl = XL + ((size_t)b * 400 + s) * 128;
  const float* xr = XR + ((size_t)b * 400 + d) * 128;
  float acc = 0.f;
  for (int j = 0; j < 4; ++j) {
    int dd = lane * 4 + j;
    float m = xl[dd] + xr[dd];
    m = m > 0.f ? m : 0.2f * m;
    acc += m * att[dd];
  }
  for (int m2 = 1; m2 < 32; m2 <<= 1) acc += __shfl_xor(acc, m2, 32);
  if (lane == 0) {
    EVAL[(size_t)b * ETOT + e] = acc;
    int* ad = (int*)&EMAX[(size_t)b * 400 + d];
    int old = *ad;
    while (__int_as_float(old) < acc) {
      int assumed = old;
      old = atomicCAS(ad, assumed, __float_as_int(acc));
      if (old == assumed) break;
    }
  }
}

__global__ void __launch_bounds__(256)
gat_edge2_kernel(float* __restrict__ EVAL, const float* __restrict__ EMAX,
                 const int* __restrict__ ei, float* __restrict__ DEN) {
  int id = blockIdx.x * 256 + threadIdx.x;
  if (id >= BB * ETOT) return;
  int b = id / ETOT, e = id % ETOT;
  int s, d; edge_sd(ei, e, s, d);
  float ex = expf(EVAL[id] - EMAX[(size_t)b * 400 + d]);
  EVAL[id] = ex;
  atomicAdd(&DEN[(size_t)b * 400 + d], ex);
}

__global__ void __launch_bounds__(256)
gat_edge3_kernel(const float* __restrict__ EVAL, const float* __restrict__ DEN,
                 const float* __restrict__ XL, const int* __restrict__ ei,
                 float* __restrict__ OUT) {
  int lane = threadIdx.x & 31;
  int id = blockIdx.x * 8 + (threadIdx.x >> 5);
  if (id >= BB * ETOT) return;
  int b = id / ETOT, e = id % ETOT;
  int s, d; edge_sd(ei, e, s, d);
  float alpha = EVAL[id] / DEN[(size_t)b * 400 + d];
  const float* xl = XL + ((size_t)b * 400 + s) * 128;
  float* out = OUT + ((size_t)b * 400 + d) * 128;
  for (int j = 0; j < 4; ++j) {
    int dd = lane * 4 + j;
    atomicAdd(&out[dd], alpha * xl[dd]);
  }
}

__global__ void fill_kernel(float* p, float v, size_t n) {
  size_t i = (size_t)blockIdx.x * 256 + threadIdx.x;
  if (i < n) p[i] = v;
}

__global__ void __launch_bounds__(256)
gather_kernel(const float* __restrict__ X, const float* __restrict__ agvs,
              const float* __restrict__ nodes, float* __restrict__ out) {
  __shared__ int ridx[4];
  int b = blockIdx.x;
  if (threadIdx.x < 4) {
    int t = threadIdx.x;
    float cx = agvs[(size_t)b * 51 * 40 + 8 + 2 * t];
    float cy = agvs[(size_t)b * 51 * 40 + 9 + 2 * t];
    int best = 0; float bd = 3.0e38f;
    for (int n = 0; n < 400; ++n) {
      float dx = cx - nodes[2 * n], dy = cy - nodes[2 * n + 1];
      float d2 = dx * dx + dy * dy;
      if (d2 < bd) { bd = d2; best = n; }
    }
    ridx[t] = best;
  }
  __syncthreads();
  for (int t = threadIdx.x; t < 512; t += 256) {
    int g = t >> 7, d = t & 127;
    out[(size_t)b * 512 + t] = X[((size_t)b * 400 + ridx[g]) * 128 + d];
  }
}

// ---------------------------------------------------------------------------
static void launch_gemm(hipStream_t st, const _Float16* A, const _Float16* W,
                        const float* bias, float* c32, _Float16* c16,
                        int M, int N, int K, bool relu) {
  dim3 g((M + 127) / 128, N / 64), b(256);
  if (relu) gemm_kernel<1><<<g, b, 0, st>>>(A, W, bias, c32, c16, M, N, K);
  else      gemm_kernel<0><<<g, b, 0, st>>>(A, W, bias, c32, c16, M, N, K);
}

static void launch_gemm_async(hipStream_t st, const _Float16* A, const _Float16* W,
                              const float* bias, float* c32, _Float16* c16,
                              int M, int N, int K, bool relu) {
  dim3 g((M + 127) / 128, N / 64), b(256);
  if (relu) gemm_async_kernel<1><<<g, b, 0, st>>>(A, W, bias, c32, c16, M, N, K);
  else      gemm_async_kernel<0><<<g, b, 0, st>>>(A, W, bias, c32, c16, M, N, K);
}

extern "C" void kernel_launch(void* const* d_in, const int* in_sizes, int n_in,
                              void* d_out, int out_size, void* d_ws, size_t ws_size,
                              hipStream_t stream)
{
  (void)in_sizes; (void)n_in; (void)out_size; (void)ws_size;
  const float* agvs  = (const float*)d_in[0];
  const float* stat  = (const float*)d_in[1];
  const float* nodes = (const float*)d_in[2];
  const int*   ei    = (const int*)d_in[3];

  // params: jax tree order (sorted dict keys), leaves 4..67
  // attn[i] (16): bk bo bq bv ffb1 ffb2 ffw1 ffw2 ln1_b ln1_g ln2_b ln2_g wk wo wq wv
  // mlp (4): b1 b2 w1 w2 ; gat (4): att bias wl wr ; gn (3): bias mean_scale weight
  const int P_ATTN0 = 4, P_ATTN1 = 20;
  const int P_EMB_AGV = 36, P_EMB_MAIN = 40, P_EMB_NODE = 44, P_EMB_STAT = 48;
  const int P_GAT0 = 52, P_GAT1 = 56, P_GN0 = 60, P_GN1 = 63;
  const int P_POSB = 66, P_POSW = 67;
  auto F32 = [&](int i) { return (const float*)d_in[i]; };

  size_t off = 0;
  auto alloc = [&](size_t bytes) -> char* {
    off = (off + 255) & ~(size_t)255;
    char* p = (char*)d_ws + off; off += bytes; return p;
  };
  auto allocW = [&](size_t n) { return (_Float16*)alloc(n * 2); };

  _Float16* ENC  = (_Float16*)alloc((size_t)16640 * 64 * 2);
  _Float16* HID  = (_Float16*)alloc((size_t)16640 * 256 * 2);
  _Float16* OBJ  = (_Float16*)alloc((size_t)16640 * 128 * 2);
  _Float16* ENCN = (_Float16*)alloc((size_t)416 * 32 * 2);
  _Float16* HIDN = (_Float16*)alloc((size_t)416 * 256 * 2);
  _Float16* NEMB = (_Float16*)alloc((size_t)416 * 128 * 2);
  const size_t NT = (size_t)BB * 400 * 128;
  float*    bufA = (float*)alloc(NT * 4);        // TGT f32 / GAT out
  _Float16* bufB = (_Float16*)alloc(NT * 2);     // TGT f16
  _Float16* bufC = (_Float16*)alloc(NT * 2);     // Q f16
  _Float16* bufD = (_Float16*)alloc(NT * 2);     // attn O f16
  float*    bufE = (float*)alloc(NT * 4);        // attn-proj / FFN out / XR
  float*    bufF = (float*)alloc(NT * 4);        // X f32 / XL
  _Float16* bufG = (_Float16*)alloc(NT * 2);     // X f16
  _Float16* bufH = (_Float16*)alloc((size_t)12800 * 512 * 2);  // FFN hidden chunk
  _Float16* K16  = (_Float16*)alloc((size_t)16640 * 128 * 2);
  _Float16* V16  = (_Float16*)alloc((size_t)16640 * 128 * 2);
  float* EVAL = (float*)alloc((size_t)BB * ETOT * 4);
  float* EMAX = (float*)alloc((size_t)BB * 400 * 4);
  float* DEN  = (float*)alloc((size_t)BB * 400 * 4);

  _Float16* WmW1 = allocW(64 * 256);  _Float16* WmW2 = allocW(256 * 128);
  _Float16* WaW1 = allocW(64 * 256);  _Float16* WaW2 = allocW(256 * 128);
  _Float16* WsW1 = allocW(64 * 256);  _Float16* WsW2 = allocW(256 * 128);
  _Float16* WnW1 = allocW(32 * 256);  _Float16* WnW2 = allocW(256 * 128);
  _Float16 *Wq[2], *Wk[2], *Wv[2], *Wo[2], *Wff1[2], *Wff2[2], *Wl[2], *Wr[2];
  for (int i = 0; i < 2; ++i) {
    Wq[i] = allocW(128 * 128); Wk[i] = allocW(128 * 128);
    Wv[i] = allocW(128 * 128); Wo[i] = allocW(128 * 128);
    Wff1[i] = allocW(128 * 512); Wff2[i] = allocW(512 * 128);
    Wl[i] = allocW(128 * 128); Wr[i] = allocW(128 * 128);
  }

  auto pack = [&](int idx, _Float16* dst, int K, int N, int Kp) {
    int tot = Kp * N;
    pack_w_kernel<<<(tot + 255) / 256, 256, 0, stream>>>(F32(idx), dst, K, N, Kp);
  };
  pack(P_EMB_MAIN + 2, WmW1, 54, 256, 64);  pack(P_EMB_MAIN + 3, WmW2, 256, 128, 256);
  pack(P_EMB_AGV + 2,  WaW1, 54, 256, 64);  pack(P_EMB_AGV + 3,  WaW2, 256, 128, 256);
  pack(P_EMB_STAT + 2, WsW1, 42, 256, 64);  pack(P_EMB_STAT + 3, WsW2, 256, 128, 256);
  pack(P_EMB_NODE + 2, WnW1, 4, 256, 32);   pack(P_EMB_NODE + 3, WnW2, 256, 128, 256);
  for (int i = 0; i < 2; ++i) {
    int pb = i ? P_ATTN1 : P_ATTN0;
    pack(pb + 14, Wq[i], 128, 128, 128); pack(pb + 12, Wk[i], 128, 128, 128);
    pack(pb + 15, Wv[i], 128, 128, 128); pack(pb + 13, Wo[i], 128, 128, 128);
    pack(pb + 6, Wff1[i], 128, 512, 128); pack(pb + 7, Wff2[i], 512, 128, 512);
    int pg = i ? P_GAT1 : P_GAT0;
    pack(pg + 2, Wl[i], 128, 128, 128); pack(pg + 3, Wr[i], 128, 128, 128);
  }

  // encode + object/node embedding MLPs (manual-staging GEMMs)
  encode_obj_kernel<<<(16640 * 64 + 255) / 256, 256, 0, stream>>>(
      agvs, stat, F32(P_POSW), F32(P_POSB), ENC);
  encode_node_kernel<<<(416 * 32 + 255) / 256, 256, 0, stream>>>(
      nodes, F32(P_POSW), F32(P_POSB), ENCN);

  launch_gemm(stream, ENC,                      WmW1, F32(P_EMB_MAIN + 0), nullptr, HID,                       256,   256, 64,  true);
  launch_gemm(stream, HID,                      WmW2, F32(P_EMB_MAIN + 1), nullptr, OBJ,                       256,   128, 256, true);
  launch_gemm(stream, ENC + (size_t)256 * 64,   WaW1, F32(P_EMB_AGV + 0),  nullptr, HID + (size_t)256 * 256,   12800, 256, 64,  true);
  launch_gemm(stream, HID + (size_t)256 * 256,  WaW2, F32(P_EMB_AGV + 1),  nullptr, OBJ + (size_t)256 * 128,   12800, 128, 256, true);
  launch_gemm(stream, ENC + (size_t)13056 * 64, WsW1, F32(P_EMB_STAT + 0), nullptr, HID + (size_t)13056 * 256, 3584,  256, 64,  true);
  launch_gemm(stream, HID + (size_t)13056 * 256,WsW2, F32(P_EMB_STAT + 1), nullptr, OBJ + (size_t)13056 * 128, 3584,  128, 256, true);
  launch_gemm(stream, ENCN, WnW1, F32(P_EMB_NODE + 0), nullptr, HIDN, 416, 256, 32,  true);
  launch_gemm(stream, HIDN, WnW2, F32(P_EMB_NODE + 1), nullptr, NEMB, 416, 128, 256, true);
  broadcast_kernel<<<(int)(NT / 256), 256, 0, stream>>>(NEMB, bufA, bufB);

  // 2 transformer decoder blocks (TDM/async-staged GEMMs)
  for (int blk = 0; blk < 2; ++blk) {
    int pb = blk ? P_ATTN1 : P_ATTN0;
    launch_gemm_async(stream, bufB, Wq[blk], F32(pb + 2), nullptr, bufC, 102400, 128, 128, false);
    launch_gemm_async(stream, OBJ,  Wk[blk], F32(pb + 0), nullptr, K16,  16640,  128, 128, false);
    launch_gemm_async(stream, OBJ,  Wv[blk], F32(pb + 3), nullptr, V16,  16640,  128, 128, false);
    attn_kernel<<<dim3(25, 256), 128, 0, stream>>>(bufC, K16, V16, agvs, stat, nodes, bufD);
    launch_gemm_async(stream, bufD, Wo[blk], F32(pb + 1), bufE, nullptr, 102400, 128, 128, false);
    add_ln_kernel<<<12800, 256, 0, stream>>>(bufA, bufE, F32(pb + 9), F32(pb + 8),
                                             bufF, bufG, 102400);
    for (int c = 0; c < 8; ++c) {              // FFN in 8 row chunks
      size_t r0 = (size_t)c * 12800;
      launch_gemm_async(stream, bufG + r0 * 128, Wff1[blk], F32(pb + 4), nullptr, bufH,
                        12800, 512, 128, true);
      launch_gemm_async(stream, bufH, Wff2[blk], F32(pb + 5), bufE + r0 * 128, nullptr,
                        12800, 128, 512, false);
    }
    add_ln_kernel<<<12800, 256, 0, stream>>>(bufF, bufE, F32(pb + 11), F32(pb + 10),
                                             bufA, bufB, 102400);
  }

  // 2 GATv2 + GraphNorm layers
  for (int gi = 0; gi < 2; ++gi) {
    int pg = gi ? P_GAT1 : P_GAT0;
    int pn = gi ? P_GN1 : P_GN0;
    launch_gemm_async(stream, bufB, Wl[gi], nullptr, bufF, nullptr, 102400, 128, 128, false);
    launch_gemm_async(stream, bufB, Wr[gi], nullptr, bufE, nullptr, 102400, 128, 128, false);
    fill_kernel<<<(102400 + 255) / 256, 256, 0, stream>>>(EMAX, -3.0e38f, (size_t)102400);
    fill_kernel<<<(102400 + 255) / 256, 256, 0, stream>>>(DEN, 0.f, (size_t)102400);
    fill_kernel<<<(int)(NT / 256), 256, 0, stream>>>(bufA, 0.f, NT);
    int nwv = BB * ETOT;
    gat_edge1_kernel<<<nwv / 8, 256, 0, stream>>>(bufF, bufE, F32(pg + 0), ei, EVAL, EMAX);
    gat_edge2_kernel<<<(nwv + 255) / 256, 256, 0, stream>>>(EVAL, EMAX, ei, DEN);
    gat_edge3_kernel<<<nwv / 8, 256, 0, stream>>>(EVAL, DEN, bufF, ei, bufA);
    graph_norm_kernel<<<256, 128, 0, stream>>>(bufA, bufB, F32(pg + 1),
                                               F32(pn + 2), F32(pn + 0), F32(pn + 1));
  }

  gather_kernel<<<256, 256, 0, stream>>>(bufA, agvs, nodes, (float*)d_out);
}
